// ArcFaceLoss_11639361372313
// MI455X (gfx1250) — matrix operand order repaired and checked by reading, not compiled, not run on hardware
//
#include <hip/hip_runtime.h>
#include <math.h>

// ---------------------------------------------------------------------------
// ArcFace loss, fused:  prep_x -> split-bf16 WMMA GEMM + online softmax
// partials -> finalize.  Never materializes the [1024 x 100000] logit matrix.
// Round 2: two batch tiles (MB=32) per iteration -> two independent WMMA
// accumulator chains, half the barriers, single masked logit-y store.
// ---------------------------------------------------------------------------

typedef __attribute__((ext_vector_type(16))) __bf16 v16bf;
typedef __attribute__((ext_vector_type(8)))  float  v8f;

#define DIM       256
#define MARGIN_M  0.1f
#define SCALE_S   1.0f
#define EPS_LS    0.1f
#define BN        128          // classes per workgroup (8 waves x 16)
#define NEG_BIG   (-1e30f)     // finite "-inf" sentinel (avoids inf-inf NaNs)

union U16bf { unsigned u[8]; v16bf v; };

__device__ __forceinline__ unsigned bf16rne(float f) {
  unsigned u = __float_as_uint(f);
  u += 0x7FFFu + ((u >> 16) & 1u);     // round-to-nearest-even to bf16
  return u >> 16;
}
__device__ __forceinline__ float bf16tof(unsigned h) {
  return __uint_as_float(h << 16);
}
// f32 pair -> packed bf16 hi pair + packed bf16 residual (lo) pair
__device__ __forceinline__ void packPair(float a, float b, unsigned& hi, unsigned& lo) {
  unsigned ha = bf16rne(a), hb = bf16rne(b);
  float ra = a - bf16tof(ha), rb = b - bf16tof(hb);
  unsigned la = bf16rne(ra), lb = bf16rne(rb);
  hi = ha | (hb << 16);
  lo = la | (lb << 16);
}

// ---------------------------------------------------------------------------
// Kernel 0: row-normalize x, emit row-major bf16 hi/lo arrays xn [m][k]
// (contiguous in k, exactly the B-operand feed layout).
// ---------------------------------------------------------------------------
__global__ __launch_bounds__(64) void prep_x_kernel(
    const float* __restrict__ x,
    unsigned short* __restrict__ xh, unsigned short* __restrict__ xl, int b) {
  const int row = blockIdx.x;
  const int tid = threadIdx.x;                       // 64 threads, 4 elems each
  const float4 v = *reinterpret_cast<const float4*>(x + (size_t)row * DIM + tid * 4);
  __shared__ float red[64];
  red[tid] = v.x * v.x + v.y * v.y + v.z * v.z + v.w * v.w;
  __syncthreads();
#pragma unroll
  for (int s = 32; s > 0; s >>= 1) {
    if (tid < s) red[tid] += red[tid + s];
    __syncthreads();
  }
  const float rn = 1.0f / fmaxf(sqrtf(red[0]), 1e-12f);
  const float vals[4] = {v.x * rn, v.y * rn, v.z * rn, v.w * rn};
#pragma unroll
  for (int j = 0; j < 4; ++j) {
    const int k = tid * 4 + j;
    const unsigned h = bf16rne(vals[j]);
    const unsigned l = bf16rne(vals[j] - bf16tof(h));
    xh[(size_t)row * DIM + k] = (unsigned short)h;
    xl[(size_t)row * DIM + k] = (unsigned short)l;
  }
}

// Per-16x16-tile epilogue: clip, margin at label, per-row max / sum-exp /
// sum-logit over this wave's 16 classes (8 per half-lane group, merged via
// shfl_xor 16).  D layout: lane = batch col (lane&15), class = cb+half*8+r.
__device__ __forceinline__ void reduce_tile(
    v8f acc, int cb, int half, int n, int m,
    const int* __restrict__ labels, float* __restrict__ logity,
    float& outM, float& outS, float& outT) {
  const int y = labels[m];
  const int rbase = cb + half * 8;
  float mx = NEG_BIG, tsum = 0.0f, ly = 0.0f;
  float logitv[8];
#pragma unroll
  for (int r = 0; r < 8; ++r) {
    const int c = rbase + r;
    float v = fminf(fmaxf(acc[r], -1.0f), 1.0f);   // clip BEFORE margin
    if (c == y) { v = SCALE_S * (v - MARGIN_M); ly = v; }
    else        { v = SCALE_S * v; }
    logitv[r] = v;
    if (c < n) { mx = fmaxf(mx, v); tsum += v; }
  }
  if ((unsigned)(y - rbase) < 8u) logity[m] = ly;   // single masked store
  float ssum = 0.0f;
#pragma unroll
  for (int r = 0; r < 8; ++r) {
    if (rbase + r < n) ssum += expf(logitv[r] - mx);
  }
  const float mxo = __shfl_xor(mx, 16, 32);
  const float sso = __shfl_xor(ssum, 16, 32);
  const float tso = __shfl_xor(tsum, 16, 32);
  outM = fmaxf(mx, mxo);
  outS = ssum * expf(mx - outM) + sso * expf(mxo - outM);
  outT = tsum + tso;
}

// ---------------------------------------------------------------------------
// Kernel 1: per workgroup = 128 classes.  A = W tile (converted f32->bf16
// hi/lo once, cached in VGPRs), B = xn^T streamed.  Split-bf16: 3 WMMAs per
// K-step per tile (ah*bh + al*bh + ah*bl), two batch tiles per iteration.
// ---------------------------------------------------------------------------
__global__ __launch_bounds__(256) void arcface_gemm_kernel(
    const float* __restrict__ W, const int* __restrict__ labels,
    const unsigned* __restrict__ xhp, const unsigned* __restrict__ xlp,
    float* __restrict__ pM, float* __restrict__ pS, float* __restrict__ pT,
    float* __restrict__ logity, int b, int n) {
  const int tid  = threadIdx.x;
  const int lane = tid & 31;
  const int wave = tid >> 5;
  const int half = lane >> 4;                    // K-half for A, class-half for D
  const int cb   = blockIdx.x * BN + wave * 16;  // this wave's class base
  const int arow = cb + (lane & 15);             // A-operand row = class index
  const bool rowValid = arow < n;
  const float* wrow = W + (size_t)(rowValid ? arow : 0) * DIM;

  // ---- load + convert this wave's W tile once (A operand, hi/lo) ----
  unsigned aHi[8][8], aLo[8][8];
#pragma unroll
  for (int kt = 0; kt < 8; ++kt) {
    const int base = kt * 32 + half * 8;         // per ISA 16-bit A layout
    float4 f0 = *reinterpret_cast<const float4*>(wrow + base);
    float4 f1 = *reinterpret_cast<const float4*>(wrow + base + 4);
    float4 f2 = *reinterpret_cast<const float4*>(wrow + base + 16);
    float4 f3 = *reinterpret_cast<const float4*>(wrow + base + 20);
    if (!rowValid) {
      f0 = make_float4(0.f, 0.f, 0.f, 0.f); f1 = f0; f2 = f0; f3 = f0;
    }
    packPair(f0.x, f0.y, aHi[kt][0], aLo[kt][0]);
    packPair(f0.z, f0.w, aHi[kt][1], aLo[kt][1]);
    packPair(f1.x, f1.y, aHi[kt][2], aLo[kt][2]);
    packPair(f1.z, f1.w, aHi[kt][3], aLo[kt][3]);
    packPair(f2.x, f2.y, aHi[kt][4], aLo[kt][4]);
    packPair(f2.z, f2.w, aHi[kt][5], aLo[kt][5]);
    packPair(f3.x, f3.y, aHi[kt][6], aLo[kt][6]);
    packPair(f3.z, f3.w, aHi[kt][7], aLo[kt][7]);
  }

  __shared__ float ldsM[8][32];
  __shared__ float ldsS[8][32];
  __shared__ float ldsT[8][32];

  for (int mb = 0; mb < b; mb += 32) {           // two 16-col batch tiles/iter
    v8f acc0 = {0.f, 0.f, 0.f, 0.f, 0.f, 0.f, 0.f, 0.f};
    v8f acc1 = {0.f, 0.f, 0.f, 0.f, 0.f, 0.f, 0.f, 0.f};
#pragma unroll
    for (int kt = 0; kt < 8; ++kt) {
      const int k = kt * 32 + half * 8;          // B operand: col = lane&15, K in table order
      const size_t boff = (size_t)(mb + (lane & 15)) * (DIM / 2) + (size_t)(k >> 1);
      const uint4 h00 = *reinterpret_cast<const uint4*>(xhp + boff);
      const uint4 h01 = *reinterpret_cast<const uint4*>(xhp + boff + 8);
      const uint4 l00 = *reinterpret_cast<const uint4*>(xlp + boff);
      const uint4 l01 = *reinterpret_cast<const uint4*>(xlp + boff + 8);
      const uint4 h10 = *reinterpret_cast<const uint4*>(xhp + boff + 16 * (DIM / 2));
      const uint4 h11 = *reinterpret_cast<const uint4*>(xhp + boff + 16 * (DIM / 2) + 8);
      const uint4 l10 = *reinterpret_cast<const uint4*>(xlp + boff + 16 * (DIM / 2));
      const uint4 l11 = *reinterpret_cast<const uint4*>(xlp + boff + 16 * (DIM / 2) + 8);
      U16bf Ah, Al, Bh0, Bl0, Bh1, Bl1;
#pragma unroll
      for (int v = 0; v < 8; ++v) { Ah.u[v] = aHi[kt][v]; Al.u[v] = aLo[kt][v]; }
      Bh0.u[0] = h00.x; Bh0.u[1] = h00.y; Bh0.u[2] = h00.z; Bh0.u[3] = h00.w;
      Bh0.u[4] = h01.x; Bh0.u[5] = h01.y; Bh0.u[6] = h01.z; Bh0.u[7] = h01.w;
      Bl0.u[0] = l00.x; Bl0.u[1] = l00.y; Bl0.u[2] = l00.z; Bl0.u[3] = l00.w;
      Bl0.u[4] = l01.x; Bl0.u[5] = l01.y; Bl0.u[6] = l01.z; Bl0.u[7] = l01.w;
      Bh1.u[0] = h10.x; Bh1.u[1] = h10.y; Bh1.u[2] = h10.z; Bh1.u[3] = h10.w;
      Bh1.u[4] = h11.x; Bh1.u[5] = h11.y; Bh1.u[6] = h11.z; Bh1.u[7] = h11.w;
      Bl1.u[0] = l10.x; Bl1.u[1] = l10.y; Bl1.u[2] = l10.z; Bl1.u[3] = l10.w;
      Bl1.u[4] = l11.x; Bl1.u[5] = l11.y; Bl1.u[6] = l11.z; Bl1.u[7] = l11.w;
      // two independent accumulation chains, interleaved
      acc0 = __builtin_amdgcn_wmma_f32_16x16x32_bf16(false, Ah.v, false, Bh0.v,
                                                     (short)0, acc0, false, false);
      acc1 = __builtin_amdgcn_wmma_f32_16x16x32_bf16(false, Ah.v, false, Bh1.v,
                                                     (short)0, acc1, false, false);
      acc0 = __builtin_amdgcn_wmma_f32_16x16x32_bf16(false, Al.v, false, Bh0.v,
                                                     (short)0, acc0, false, false);
      acc1 = __builtin_amdgcn_wmma_f32_16x16x32_bf16(false, Al.v, false, Bh1.v,
                                                     (short)0, acc1, false, false);
      acc0 = __builtin_amdgcn_wmma_f32_16x16x32_bf16(false, Ah.v, false, Bl0.v,
                                                     (short)0, acc0, false, false);
      acc1 = __builtin_amdgcn_wmma_f32_16x16x32_bf16(false, Ah.v, false, Bl1.v,
                                                     (short)0, acc1, false, false);
    }

    float M0, S0, T0, M1, S1, T1;
    reduce_tile(acc0, cb, half, n, mb + (lane & 15), labels, logity, M0, S0, T0);
    reduce_tile(acc1, cb, half, n, mb + 16 + (lane & 15), labels, logity, M1, S1, T1);
    if (lane < 16) {
      ldsM[wave][lane] = M0;      ldsS[wave][lane] = S0;      ldsT[wave][lane] = T0;
      ldsM[wave][lane + 16] = M1; ldsS[wave][lane + 16] = S1; ldsT[wave][lane + 16] = T1;
    }
    __syncthreads();
    if (tid < 32) {                              // fixed-order 8-wave merge
      float M = NEG_BIG, S = 0.0f, T = 0.0f;
#pragma unroll
      for (int w = 0; w < 8; ++w) {
        const float bm = ldsM[w][tid], bs = ldsS[w][tid];
        const float m2 = fmaxf(M, bm);
        S = S * expf(M - m2) + bs * expf(bm - m2);
        M = m2;
        T += ldsT[w][tid];
      }
      const size_t o = (size_t)blockIdx.x * b + (size_t)(mb + tid);
      pM[o] = M; pS[o] = S; pT[o] = T;
    }
    __syncthreads();
  }
}

// ---------------------------------------------------------------------------
// Kernel 2: per row combine block partials (fixed order), lse, row loss,
// tree-reduce 1024 rows to the scalar loss.
// ---------------------------------------------------------------------------
__global__ __launch_bounds__(1024) void finalize_kernel(
    const float* __restrict__ pM, const float* __restrict__ pS,
    const float* __restrict__ pT, const float* __restrict__ logity,
    float* __restrict__ out, int b, int n, int nb) {
  const int m = threadIdx.x;
  float M = NEG_BIG, S = 0.0f, T = 0.0f;
  for (int blk = 0; blk < nb; ++blk) {
    const size_t o = (size_t)blk * b + m;
    const float bm = pM[o], bs = pS[o], bt = pT[o];
    const float m2 = fmaxf(M, bm);
    S = S * expf(M - m2) + bs * expf(bm - m2);
    M = m2; T += bt;
  }
  const float lse = M + logf(S);
  const float rl  = lse - (1.0f - EPS_LS) * logity[m] - (EPS_LS / (float)n) * T;
  __shared__ float red[1024];
  red[m] = rl;
  __syncthreads();
#pragma unroll
  for (int s = 512; s > 0; s >>= 1) {
    if (m < s) red[m] += red[m + s];
    __syncthreads();
  }
  if (m == 0) out[0] = red[0] / (float)b;
}

// ---------------------------------------------------------------------------
extern "C" void kernel_launch(void* const* d_in, const int* in_sizes, int n_in,
                              void* d_out, int out_size, void* d_ws, size_t ws_size,
                              hipStream_t stream) {
  const float* x      = (const float*)d_in[0];   // [b, 256] f32
  const float* W      = (const float*)d_in[1];   // [n, 256] f32, unit rows
  const int*   labels = (const int*)d_in[2];     // [b] i32
  const int b  = in_sizes[2];                    // 1024
  const int n  = in_sizes[1] / DIM;              // 100000
  const int nb = (n + BN - 1) / BN;              // 782 class blocks

  // workspace carve-up (~10.5 MB total)
  char* ws = (char*)d_ws;
  size_t off = 0;
  unsigned short* xh = (unsigned short*)(ws + off); off += (size_t)DIM * b * 2;
  unsigned short* xl = (unsigned short*)(ws + off); off += (size_t)DIM * b * 2;
  float* logity = (float*)(ws + off); off += (size_t)b * 4;
  off = (off + 255) & ~(size_t)255;
  float* pM = (float*)(ws + off); off += (size_t)nb * b * 4;
  float* pS = (float*)(ws + off); off += (size_t)nb * b * 4;
  float* pT = (float*)(ws + off); off += (size_t)nb * b * 4;

  prep_x_kernel<<<b, 64, 0, stream>>>(x, xh, xl, b);
  arcface_gemm_kernel<<<nb, 256, 0, stream>>>(
      W, labels, (const unsigned*)xh, (const unsigned*)xl,
      pM, pS, pT, logity, b, n);
  finalize_kernel<<<1, b, 0, stream>>>(pM, pS, pT, logity, (float*)d_out, b, n, nb);
}
